// CausalMultiHeadSelfAttention_30485677867606
// MI455X (gfx1250) — matrix operand order repaired and checked by reading, not compiled, not run on hardware
//
#include <hip/hip_runtime.h>
#include <hip/hip_bf16.h>

#define B_   2
#define S_   2048
#define D_   1024
#define H_   16
#define HKV_ 4
#define HD_  64
#define NQKV 1536   // 1024 (Q) + 256 (K) + 256 (V)

typedef __attribute__((ext_vector_type(16))) _Float16 v16h;
typedef __attribute__((ext_vector_type(8)))  _Float16 v8h;
typedef __attribute__((ext_vector_type(8)))  float    v8f;

// ---------------------------------------------------------------------------
// WMMA fragment helpers (layouts per CDNA5 ISA §7.12.2, wave32)
// ---------------------------------------------------------------------------
__device__ __forceinline__ v16h cat16(v8h lo, v8h hi) {
  return __builtin_shufflevector(lo, hi, 0,1,2,3,4,5,6,7,8,9,10,11,12,13,14,15);
}

// A-matrix 16x32 f16, row-major source with leading dim `ld` (halves).
// lane m = l&15, kh = l>>4 ; elems 0..7 -> K = 8*kh + 0..7 ; 8..15 -> K = 16 + 8*kh + 0..7
__device__ __forceinline__ v16h load_frag_A(const _Float16* p, int ld) {
  const int lane = threadIdx.x & 31;
  const int m = lane & 15, kh = lane >> 4;
  const _Float16* r = p + (size_t)m * ld + kh * 8;
  v8h lo = *(const v8h*)r;
  v8h hi = *(const v8h*)(r + 16);
  return cat16(lo, hi);
}

// B-matrix 32x16 f16 supplied as B^T rows: element (kd, n) read from p[n*ld + kd].
// lane n = l&15, kg = l>>4 ; elems 0..7 -> kd = 8*kg + 0..7 ; 8..15 -> kd = 16 + 8*kg + 0..7
__device__ __forceinline__ v16h load_frag_B(const _Float16* p, int ld) {
  const int lane = threadIdx.x & 31;
  const int n = lane & 15, kg = lane >> 4;
  const _Float16* r = p + (size_t)n * ld + kg * 8;
  v8h lo = *(const v8h*)r;
  v8h hi = *(const v8h*)(r + 16);
  return cat16(lo, hi);
}

__device__ __forceinline__ v8f wmma16(v16h a, v16h b, v8f c) {
  return __builtin_amdgcn_wmma_f32_16x16x32_f16(false, a, false, b, (short)0, c,
                                                false, false);
}

// ---------------------------------------------------------------------------
// Generic GEMM: C[M,N] = A[M,K] (f16, row-major) x BT[N,K]^T (f16) -> f32
// One wave computes a 32x64 tile (2x4 accumulators).
// ---------------------------------------------------------------------------
__global__ __launch_bounds__(32) void gemm_f16_tn(const _Float16* __restrict__ A,
                                                  const _Float16* __restrict__ BT,
                                                  float* __restrict__ C,
                                                  int M, int N, int K) {
  const int n0 = blockIdx.x * 64;
  const int m0 = blockIdx.y * 32;
  const int lane = threadIdx.x & 31;
  const int n = lane & 15, kh = lane >> 4;

  v8f zero = {};
  v8f acc[2][4];
#pragma unroll
  for (int i = 0; i < 2; ++i)
#pragma unroll
    for (int t = 0; t < 4; ++t) acc[i][t] = zero;

  for (int kk = 0; kk < K; kk += 32) {
    __builtin_prefetch(A + (size_t)m0 * K + kk + 128, 0, 1);
    __builtin_prefetch(BT + (size_t)n0 * K + kk + 128, 0, 1);
    v16h a0 = load_frag_A(A + (size_t)m0 * K + kk, K);
    v16h a1 = load_frag_A(A + (size_t)(m0 + 16) * K + kk, K);
    v16h b0 = load_frag_B(BT + (size_t)(n0 +  0) * K + kk, K);
    v16h b1 = load_frag_B(BT + (size_t)(n0 + 16) * K + kk, K);
    v16h b2 = load_frag_B(BT + (size_t)(n0 + 32) * K + kk, K);
    v16h b3 = load_frag_B(BT + (size_t)(n0 + 48) * K + kk, K);
    acc[0][0] = wmma16(a0, b0, acc[0][0]);
    acc[0][1] = wmma16(a0, b1, acc[0][1]);
    acc[0][2] = wmma16(a0, b2, acc[0][2]);
    acc[0][3] = wmma16(a0, b3, acc[0][3]);
    acc[1][0] = wmma16(a1, b0, acc[1][0]);
    acc[1][1] = wmma16(a1, b1, acc[1][1]);
    acc[1][2] = wmma16(a1, b2, acc[1][2]);
    acc[1][3] = wmma16(a1, b3, acc[1][3]);
  }
  // C/D layout: vgpr j -> row j + 8*kh, col = lane&15
#pragma unroll
  for (int i = 0; i < 2; ++i)
#pragma unroll
    for (int t = 0; t < 4; ++t)
#pragma unroll
      for (int j = 0; j < 8; ++j)
        C[(size_t)(m0 + 16 * i + j + 8 * kh) * N + (n0 + 16 * t + n)] = acc[i][t][j];
}

// ---------------------------------------------------------------------------
// Elementwise conversions / weight packing
// ---------------------------------------------------------------------------
__global__ void cvt_f32_to_f16(const float* __restrict__ s, _Float16* __restrict__ d,
                               int nElem) {
  int i = blockIdx.x * blockDim.x + threadIdx.x;
  if (i < nElem) d[i] = (_Float16)s[i];
}

// wqkvT[n][k]: n<1024 -> wq[k][n]; 1024<=n<1280 -> wk[k][n-1024]; else wv[k][n-1280]
__global__ void pack_wqkv(const float* __restrict__ wq, const float* __restrict__ wk,
                          const float* __restrict__ wv, _Float16* __restrict__ wqkvT) {
  const int nn = blockIdx.x;  // 0..1535
  for (int k = threadIdx.x; k < D_; k += blockDim.x) {
    float v;
    if (nn < 1024)      v = wq[(size_t)k * 1024 + nn];
    else if (nn < 1280) v = wk[(size_t)k * 256 + (nn - 1024)];
    else                v = wv[(size_t)k * 256 + (nn - 1280)];
    wqkvT[(size_t)nn * D_ + k] = (_Float16)v;
  }
}

__global__ void pack_wo(const float* __restrict__ wo, _Float16* __restrict__ woT) {
  const int nn = blockIdx.x;  // 0..1023
  for (int k = threadIdx.x; k < D_; k += blockDim.x)
    woT[(size_t)nn * D_ + k] = (_Float16)wo[(size_t)k * D_ + nn];
}

// ---------------------------------------------------------------------------
// RMSNorm + RoPE (Q,K) and V transpose, f32 -> f16.
// grid (S, B, H + HKV + HKV), block 32. Lane i handles rope pair (i, i+32).
// ---------------------------------------------------------------------------
__global__ __launch_bounds__(32) void norm_rope_kernel(const float* __restrict__ qkv,
                                                       _Float16* __restrict__ q16,
                                                       _Float16* __restrict__ k16,
                                                       _Float16* __restrict__ vT16) {
  const int lane = threadIdx.x & 31;
  const int s = blockIdx.x, b = blockIdx.y, hh = blockIdx.z;
  const float* row = qkv + ((size_t)b * S_ + s) * NQKV;

  if (hh >= H_ + HKV_) {  // V: straight transpose-convert
    const int kvh = hh - (H_ + HKV_);
    const float* src = row + 1024 + 256 + kvh * HD_;
    vT16[((size_t)(b * HKV_ + kvh) * HD_ + lane) * S_ + s]        = (_Float16)src[lane];
    vT16[((size_t)(b * HKV_ + kvh) * HD_ + lane + 32) * S_ + s]   = (_Float16)src[lane + 32];
    return;
  }

  const float* src;
  _Float16* dst;
  if (hh < H_) {
    src = row + hh * HD_;
    dst = q16 + ((size_t)(b * H_ + hh) * S_ + s) * HD_;
  } else {
    const int kvh = hh - H_;
    src = row + 1024 + kvh * HD_;
    dst = k16 + ((size_t)(b * HKV_ + kvh) * S_ + s) * HD_;
  }
  float x1 = src[lane], x2 = src[lane + 32];
  float ssq = x1 * x1 + x2 * x2;
#pragma unroll
  for (int off = 16; off >= 1; off >>= 1) ssq += __shfl_xor(ssq, off, 32);
  const float r = rsqrtf(ssq * (1.0f / 64.0f) + 1e-5f);
  x1 *= r; x2 *= r;
  const float inv = __powf(10000.0f, -(float)lane * (1.0f / 32.0f));
  const float ang = (float)s * inv;
  float sn, cs;
  sincosf(ang, &sn, &cs);
  dst[lane]      = (_Float16)(x1 * cs - x2 * sn);
  dst[lane + 32] = (_Float16)(x2 * cs + x1 * sn);
}

// ---------------------------------------------------------------------------
// Flash attention: one wave per (16-query tile, head, batch).
// Scores 2xWMMA per 16-key tile, tanh softcap, causal mask, online softmax,
// P staged through LDS (C-layout -> A-layout), PV over 32-key blocks.
// ---------------------------------------------------------------------------
__global__ __launch_bounds__(32) void attn_kernel(const _Float16* __restrict__ q16,
                                                  const _Float16* __restrict__ k16,
                                                  const _Float16* __restrict__ vT16,
                                                  _Float16* __restrict__ y16) {
  __shared__ _Float16 lds_p[16 * 32];
  const int lane = threadIdx.x & 31;
  const int n = lane & 15, kh = lane >> 4;
  const int q0 = blockIdx.x * 16;
  const int h = blockIdx.y;
  const int b = blockIdx.z;
  const int kv = h >> 2;  // repeat factor H/HKV = 4

  const _Float16* qptr = q16 + ((size_t)(b * H_ + h) * S_ + q0) * HD_;
  const v16h aq0 = load_frag_A(qptr, HD_);         // d 0..31
  const v16h aq1 = load_frag_A(qptr + 32, HD_);    // d 32..63
  const _Float16* kbase = k16 + (size_t)(b * HKV_ + kv) * S_ * HD_;
  const _Float16* vbase = vT16 + (size_t)(b * HKV_ + kv) * HD_ * S_;

  v8f zero = {};
  float m_r[8], l_r[8];
  v8f acc[4];
#pragma unroll
  for (int j = 0; j < 8; ++j) { m_r[j] = -1e30f; l_r[j] = 0.0f; }
#pragma unroll
  for (int t = 0; t < 4; ++t) acc[t] = zero;

  for (int kb = 0; kb <= q0 + 15; kb += 32) {
    const bool valid1 = (kb + 16) <= (q0 + 15);
    v8f sc0, sc1 = zero;
    {
      v16h b0 = load_frag_B(kbase + (size_t)kb * HD_, HD_);
      v16h b1 = load_frag_B(kbase + (size_t)kb * HD_ + 32, HD_);
      sc0 = wmma16(aq0, b0, zero);
      sc0 = wmma16(aq1, b1, sc0);
    }
    if (valid1) {
      v16h b0 = load_frag_B(kbase + (size_t)(kb + 16) * HD_, HD_);
      v16h b1 = load_frag_B(kbase + (size_t)(kb + 16) * HD_ + 32, HD_);
      sc1 = wmma16(aq0, b0, zero);
      sc1 = wmma16(aq1, b1, sc1);
    }
#pragma unroll
    for (int j = 0; j < 8; ++j) {
      const int qi = q0 + j + 8 * kh;
      float s0 = sc0[j] * 0.125f;                 // / sqrt(64)
      s0 = tanhf(s0 * 0.02f) * 50.0f;             // softcap
      if (kb + n > qi) s0 = -1e30f;               // causal
      float s1 = -1e30f;
      if (valid1) {
        s1 = sc1[j] * 0.125f;
        s1 = tanhf(s1 * 0.02f) * 50.0f;
        if (kb + 16 + n > qi) s1 = -1e30f;
      }
      float rm = fmaxf(s0, s1);
#pragma unroll
      for (int off = 8; off >= 1; off >>= 1) rm = fmaxf(rm, __shfl_xor(rm, off, 16));
      const float mn = fmaxf(m_r[j], rm);
      const float p0 = __expf(s0 - mn);
      const float p1 = valid1 ? __expf(s1 - mn) : 0.0f;
      float rs = p0 + p1;
#pragma unroll
      for (int off = 8; off >= 1; off >>= 1) rs += __shfl_xor(rs, off, 16);
      const float corr = __expf(m_r[j] - mn);
      l_r[j] = l_r[j] * corr + rs;
      m_r[j] = mn;
#pragma unroll
      for (int t = 0; t < 4; ++t) acc[t][j] *= corr;
      lds_p[(j + 8 * kh) * 32 + n]      = (_Float16)p0;
      lds_p[(j + 8 * kh) * 32 + 16 + n] = (_Float16)p1;
    }
    __syncthreads();
    const v16h ap = load_frag_A(lds_p, 32);       // P 16x32 (queries x keys)
#pragma unroll
    for (int t = 0; t < 4; ++t) {
      v16h bv = load_frag_B(vbase + (size_t)(16 * t) * S_ + kb, S_);  // V^T rows
      acc[t] = wmma16(ap, bv, acc[t]);
    }
    __syncthreads();
  }
#pragma unroll
  for (int t = 0; t < 4; ++t)
#pragma unroll
    for (int j = 0; j < 8; ++j) {
      const size_t row = (size_t)(b * S_ + q0 + j + 8 * kh);
      y16[row * (size_t)(H_ * HD_) + h * HD_ + 16 * t + n] =
          (_Float16)(acc[t][j] / l_r[j]);
    }
}

// ---------------------------------------------------------------------------
// Host-side launch
// ---------------------------------------------------------------------------
extern "C" void kernel_launch(void* const* d_in, const int* in_sizes, int n_in,
                              void* d_out, int out_size, void* d_ws, size_t ws_size,
                              hipStream_t stream) {
  (void)in_sizes; (void)n_in; (void)out_size; (void)ws_size;
  const float* x  = (const float*)d_in[0];
  const float* wq = (const float*)d_in[1];
  const float* wk = (const float*)d_in[2];
  const float* wv = (const float*)d_in[3];
  const float* wo = (const float*)d_in[4];
  float* out = (float*)d_out;

  char* ws = (char*)d_ws;
  size_t off = 0;
  _Float16* x16   = (_Float16*)(ws + off); off += (size_t)B_ * S_ * D_ * 2;        // 8 MB
  _Float16* wqkvT = (_Float16*)(ws + off); off += (size_t)NQKV * D_ * 2;           // 3 MB
  _Float16* woT   = (_Float16*)(ws + off); off += (size_t)D_ * D_ * 2;             // 2 MB
  float*    qkv   = (float*)   (ws + off); off += (size_t)B_ * S_ * NQKV * 4;      // 24 MB
  _Float16* q16   = (_Float16*)(ws + off); off += (size_t)B_ * H_ * S_ * HD_ * 2;  // 8 MB
  _Float16* k16   = (_Float16*)(ws + off); off += (size_t)B_ * HKV_ * S_ * HD_ * 2;// 2 MB
  _Float16* vT16  = (_Float16*)(ws + off); off += (size_t)B_ * HKV_ * HD_ * S_ * 2;// 2 MB
  _Float16* y16   = (_Float16*)(ws + off); off += (size_t)B_ * S_ * D_ * 2;        // 8 MB

  const int nx = B_ * S_ * D_;
  cvt_f32_to_f16<<<(nx + 255) / 256, 256, 0, stream>>>(x, x16, nx);
  pack_wqkv<<<NQKV, 256, 0, stream>>>(wq, wk, wv, wqkvT);
  pack_wo<<<D_, 256, 0, stream>>>(wo, woT);

  // QKV projection: [4096 x 1536] = x16 [4096 x 1024] @ wqkvT^T
  gemm_f16_tn<<<dim3(NQKV / 64, (B_ * S_) / 32), 32, 0, stream>>>(
      x16, wqkvT, qkv, B_ * S_, NQKV, D_);

  norm_rope_kernel<<<dim3(S_, B_, H_ + 2 * HKV_), 32, 0, stream>>>(qkv, q16, k16, vT16);

  attn_kernel<<<dim3(S_ / 16, H_, B_), 32, 0, stream>>>(q16, k16, vT16, y16);

  // Output projection: out [4096 x 1024] = y16 @ woT^T
  gemm_f16_tn<<<dim3(D_ / 64, (B_ * S_) / 32), 32, 0, stream>>>(
      y16, woT, out, B_ * S_, D_, D_);
}
